// DeepseekMoE_42236708389026
// MI455X (gfx1250) — matrix operand order repaired and checked by reading, not compiled, not run on hardware
//
#include <hip/hip_runtime.h>

// ---------------------------------------------------------------------------
// DeepSeek-style MoE for MI455X (gfx1250, wave32, WMMA + Tensor Data Mover).
// T=4096 tokens, H=1024, E=64 experts (I=704), top-8 routing, 2 shared experts.
// ---------------------------------------------------------------------------

#define T_TOK 4096
#define HDIM  1024
#define EXP   64
#define IDIM  704
#define XS_STR 1032   // HDIM + 8 halves -> 2064B rows, 16B aligned, conflict-free
#define HS_STR 712    // IDIM + 8 halves -> 1424B rows, 16B aligned

// Dynamic-LDS layout (byte offsets; no static LDS in this kernel -> offset 0 base)
#define LDS_TS    0                      // int[16]
#define LDS_WT    64                     // float[16]
#define LDS_XS    128                    // f16[16*XS_STR] = 33024 B
#define LDS_HS    (128 + 33024)          // f16[16*HS_STR] = 22784 B
#define LDS_SLOTS (128 + 33024 + 22784)  // 8 waves * 8192 B TDM slots (64KB)
                                         // (also reused as 16x1024 f32 X-gather staging)
#define LDS_TOTAL (LDS_SLOTS + 8 * 8192)

typedef _Float16 f16;
typedef f16   v16h __attribute__((ext_vector_type(16)));
typedef f16   v8h  __attribute__((ext_vector_type(8)));
typedef float v8f  __attribute__((ext_vector_type(8)));
typedef unsigned int u32x4 __attribute__((ext_vector_type(4)));
typedef int i32x4 __attribute__((ext_vector_type(4)));
typedef int i32x8 __attribute__((ext_vector_type(8)));

static __device__ __forceinline__ v16h concat8(v8h lo, v8h hi) {
    return __builtin_shufflevector(lo, hi, 0,1,2,3,4,5,6,7,8,9,10,11,12,13,14,15);
}

#if __has_builtin(__builtin_amdgcn_tensor_load_to_lds)
#define USE_TDM 1
#else
#define USE_TDM 0
#endif

#if __has_builtin(__builtin_amdgcn_s_wait_tensorcnt)
#define TDM_WAIT(n) do { __builtin_amdgcn_s_wait_tensorcnt(n); \
                         __asm__ volatile("" ::: "memory"); } while (0)
#else
#define TDM_WAIT(n) do { __asm__ volatile("" ::: "memory"); } while (0)
#endif

#if USE_TDM
static __device__ __forceinline__ void tdm_issue(u32x4 g0, i32x8 g1, i32x4 g2, i32x4 g3) {
#if defined(__clang_major__) && (__clang_major__ >= 23)
    i32x8 z8 = {0, 0, 0, 0, 0, 0, 0, 0};
    __builtin_amdgcn_tensor_load_to_lds(g0, g1, g2, g3, z8, 0);
#else
    __builtin_amdgcn_tensor_load_to_lds(g0, g1, g2, g3, 0);
#endif
}

// 2D fp32 tile: tile_rows x tile_cols, row pitch = stride_elems, packed into LDS.
static __device__ __forceinline__ void tdm_load_tile_f32(
    unsigned lds_off, const float* gptr, unsigned stride_elems,
    unsigned tile_rows, unsigned tile_cols)
{
    unsigned long long ga = (unsigned long long)(size_t)gptr;
    u32x4 g0;
    g0.x = 1u;                                              // count=1, user mode
    g0.y = lds_off;                                         // lds_addr
    g0.z = (unsigned)ga;                                    // global_addr[31:0]
    g0.w = (unsigned)((ga >> 32) & 0x1FFFFFFu) | (2u << 30);// addr[56:32] | type=2
    i32x8 g1;
    const unsigned td0 = tile_cols, td1 = tile_rows;        // exact-fit tensor dims
    g1[0] = (int)(2u << 16);                                // data_size = 4B
    g1[1] = (int)((td0 & 0xFFFFu) << 16);                   // tensor_dim0[15:0]
    g1[2] = (int)((td0 >> 16) | ((td1 & 0xFFFFu) << 16));   // dim0 hi | dim1 lo
    g1[3] = (int)((td1 >> 16) | (tile_cols << 16));         // dim1 hi | tile_dim0
    g1[4] = (int)tile_rows;                                 // tile_dim1 (tile_dim2=0)
    g1[5] = (int)stride_elems;                              // tensor_dim0_stride lo
    g1[6] = 0; g1[7] = 0;
    i32x4 z4 = {0, 0, 0, 0};
    tdm_issue(g0, g1, z4, z4);
}

// Gather mode: 16 full rows of `row_elems` fp32, 16-bit row indices.
static __device__ __forceinline__ void tdm_gather_rows_f32(
    unsigned lds_off, const float* base, unsigned row_elems,
    unsigned tensor_rows, const int* idx)
{
    unsigned long long ga = (unsigned long long)(size_t)base;
    u32x4 g0;
    g0.x = 1u | (1u << 31);                                 // gather_mode, 16b idx
    g0.y = lds_off;
    g0.z = (unsigned)ga;
    g0.w = (unsigned)((ga >> 32) & 0x1FFFFFFu) | (2u << 30);
    i32x8 g1;
    const unsigned td0 = row_elems, td1 = tensor_rows;
    g1[0] = (int)(2u << 16);
    g1[1] = (int)((td0 & 0xFFFFu) << 16);
    g1[2] = (int)((td0 >> 16) | ((td1 & 0xFFFFu) << 16));
    g1[3] = (int)((td1 >> 16) | (row_elems << 16));         // tile_dim0 = row_elems
    g1[4] = 16;                                             // tile_dim1 = #indices
    g1[5] = (int)row_elems;
    g1[6] = 0; g1[7] = 0;
    i32x4 g2, g3;
    g2[0] = (idx[0]  & 0xFFFF) | (idx[1]  << 16);
    g2[1] = (idx[2]  & 0xFFFF) | (idx[3]  << 16);
    g2[2] = (idx[4]  & 0xFFFF) | (idx[5]  << 16);
    g2[3] = (idx[6]  & 0xFFFF) | (idx[7]  << 16);
    g3[0] = (idx[8]  & 0xFFFF) | (idx[9]  << 16);
    g3[1] = (idx[10] & 0xFFFF) | (idx[11] << 16);
    g3[2] = (idx[12] & 0xFFFF) | (idx[13] << 16);
    g3[3] = (idx[14] & 0xFFFF) | (idx[15] << 16);
    tdm_issue(g0, g1, g2, g3);
}
#endif // USE_TDM

// ---------------------------------------------------------------------------
// Router: one wave per token; grouped top-k; append (token, weight) to
// per-expert gather lists.
// ---------------------------------------------------------------------------
__global__ __launch_bounds__(256) void router_kernel(
    const float* __restrict__ x, const float* __restrict__ gate_w,
    const float* __restrict__ gate_bias, int* __restrict__ cnt,
    int* __restrict__ exp_tok, float* __restrict__ exp_w)
{
    __shared__ float sc_s[8][64];   // biased scores (selection)
    __shared__ float sco_s[8][64];  // raw sigmoid scores (weights)

    const int tid  = threadIdx.x;
    const int lane = tid & 31;
    const int wv   = tid >> 5;
    const int t    = blockIdx.x * 8 + wv;

    const float4* xr = (const float4*)(x + (size_t)t * HDIM);
    const float4* g0 = (const float4*)(gate_w + (size_t)lane * HDIM);
    const float4* g1 = (const float4*)(gate_w + (size_t)(lane + 32) * HDIM);
    float a0 = 0.f, a1 = 0.f;
    for (int i = 0; i < HDIM / 4; ++i) {
        float4 xv = xr[i], w0 = g0[i], w1 = g1[i];
        a0 += xv.x*w0.x + xv.y*w0.y + xv.z*w0.z + xv.w*w0.w;
        a1 += xv.x*w1.x + xv.y*w1.y + xv.z*w1.z + xv.w*w1.w;
    }
    float s0 = __builtin_amdgcn_rcpf(1.f + __expf(-a0));
    float s1 = __builtin_amdgcn_rcpf(1.f + __expf(-a1));
    sco_s[wv][lane]      = s0;
    sco_s[wv][lane + 32] = s1;
    sc_s[wv][lane]       = s0 + gate_bias[lane];
    sc_s[wv][lane + 32]  = s1 + gate_bias[lane + 32];
    __syncthreads();

    if (tid < 8) {
        const int    tk  = blockIdx.x * 8 + tid;
        const float* sc  = sc_s[tid];
        const float* sco = sco_s[tid];
        float gs[8];
        for (int g = 0; g < 8; ++g) {
            float m1 = -1e30f, m2 = -1e30f;
            for (int j = 0; j < 8; ++j) {
                float v = sc[g * 8 + j];
                if (v > m1) { m2 = m1; m1 = v; } else if (v > m2) { m2 = v; }
            }
            gs[g] = m1 + m2;
        }
        unsigned gmask = 0;
        for (int it = 0; it < 4; ++it) {
            float best = -1e30f; int bi = 0;
            for (int g = 0; g < 8; ++g)
                if (!((gmask >> g) & 1u) && gs[g] > best) { best = gs[g]; bi = g; }
            gmask |= 1u << bi;
        }
        unsigned long long taken = 0;
        int eidx[8]; float ew[8]; float wsum = 0.f;
        for (int k = 0; k < 8; ++k) {
            float best = -1e30f; int bi = 0;
            for (int e = 0; e < 64; ++e) {
                if (((gmask >> (e >> 3)) & 1u) && !((taken >> e) & 1ull)) {
                    float v = sc[e];
                    if (v > best) { best = v; bi = e; }
                }
            }
            taken |= 1ull << bi;
            eidx[k] = bi; ew[k] = sco[bi]; wsum += sco[bi];
        }
        const float scale = 2.5f / wsum;
        for (int k = 0; k < 8; ++k) {
            int e = eidx[k];
            int slot = atomicAdd(&cnt[e], 1);
            exp_tok[e * T_TOK + slot] = tk;
            exp_w  [e * T_TOK + slot] = ew[k] * scale;
        }
    }
}

// ---------------------------------------------------------------------------
// Fused expert MLP: out[t] += w_t * ( silu(x@Wg) * (x@Wu) ) @ Wd
// grid.y in [0,64): routed expert e; grid.y in [64,66): shared column-half.
// Block = 256 threads (8 waves), tile = 16 gathered tokens.
// ---------------------------------------------------------------------------
__global__ __launch_bounds__(256) void moe_expert_kernel(
    const float* __restrict__ x,
    const float* __restrict__ w_gate, const float* __restrict__ w_up,
    const float* __restrict__ w_down,
    const float* __restrict__ ws_gate, const float* __restrict__ ws_up,
    const float* __restrict__ ws_down,
    const int* __restrict__ cnt, const int* __restrict__ exp_tok,
    const float* __restrict__ exp_w, float* __restrict__ out)
{
    const int e = blockIdx.y;
    const bool shared_mode = (e >= EXP);

    const float *pg, *pu, *pd;
    int ldgu, count;
    if (shared_mode) {
        const int s = e - EXP;
        pg = ws_gate + (size_t)s * IDIM;                // [H, 2I] column half
        pu = ws_up   + (size_t)s * IDIM;
        pd = ws_down + (size_t)s * IDIM * HDIM;         // [2I, H] row half
        ldgu = 2 * IDIM;
        count = T_TOK;
    } else {
        const size_t off = (size_t)e * HDIM * IDIM;
        pg = w_gate + off;                              // [H, I]
        pu = w_up   + off;
        pd = w_down + (size_t)e * IDIM * HDIM;          // [I, H]
        ldgu = IDIM;
        count = cnt[e];
    }
    const int m0 = blockIdx.x * 16;
    if (m0 >= count) return;
    const int rowsValid = (count - m0 < 16) ? (count - m0) : 16;

    extern __shared__ char smem[];
    int*   Ts   = (int*)  (smem + LDS_TS);
    float* Wt   = (float*)(smem + LDS_WT);
    f16*   Xs   = (f16*)  (smem + LDS_XS);
    f16*   Hs   = (f16*)  (smem + LDS_HS);
    float* SLOT = (float*)(smem + LDS_SLOTS);   // single AS(3) base, int indexing

    const int tid  = threadIdx.x;
    const int lane = tid & 31;
    const int wv   = tid >> 5;

    if (tid < 16) {
        int gi = m0 + tid;
        int t = 0; float wgt = 0.f;
        if (shared_mode) { t = gi; wgt = 1.f; }
        else if (gi < count) { t = exp_tok[e * T_TOK + gi]; wgt = exp_w[e * T_TOK + gi]; }
        Ts[tid] = t; Wt[tid] = wgt;
    }
    __syncthreads();

    // ---- Stage 1: gather 16 token rows of x -> LDS f16 ----------------------
#if USE_TDM
    if (wv == 0) {   // one TDM gather pulls all 16 fp32 rows into the slot region
        int idx[16];
        #pragma unroll
        for (int r = 0; r < 16; ++r) idx[r] = Ts[r];
        tdm_gather_rows_f32(LDS_SLOTS, x, HDIM, T_TOK, idx);
        TDM_WAIT(0);
    }
    __syncthreads();
    for (int i = tid; i < 16 * HDIM; i += 256) {
        const int r = i >> 10, c = i & (HDIM - 1);
        Xs[r * XS_STR + c] = (f16)SLOT[i];
    }
#else
    for (int i = tid; i < 16 * HDIM; i += 256) {
        const int r = i >> 10, c = i & (HDIM - 1);
        Xs[r * XS_STR + c] = (f16)x[(size_t)Ts[r] * HDIM + c];
    }
#endif
    __syncthreads();

    // Operand k layout used here, contiguous, same in A and B (f16 16x16x32):
    //  A: lanes 0-15 hold K 0..15; lanes 16-31 hold K 16..31
    //  B: lanes 0-15 hold K 0..15; lanes 16-31 hold K 16..31
    const int nIdx = lane & 15;
    const int kA   = (lane & 16) ? 8 : 0;
    const int kB   = (lane & 16) ? 16 : 0;

#if USE_TDM
    // per-wave double-buffered 32Kx16N fp32 panels:
    //   floats: wave region = 2048; G buffers at +0/+512, U buffers at +1024/+1536
    const unsigned slotGoff = LDS_SLOTS + (unsigned)wv * 8192;  // bytes, for TDM
    const unsigned slotUoff = slotGoff + 4096;
    const int gBase0 = wv * 2048;          // float index of G buffers
    const int uBase0 = gBase0 + 1024;      // float index of U buffers
#endif

    // ---- Stage 2: C1 = X@Wg, C2 = X@Wu; h = silu(C1)*C2 -> Hs ---------------
    for (int nt = wv; nt < IDIM / 16; nt += 8) {
        const int n0 = nt * 16;
        v8f c1 = {}, c2 = {};
        const f16* arow = Xs + (lane & 15) * XS_STR;
#if USE_TDM
        tdm_load_tile_f32(slotGoff, pg + n0, (unsigned)ldgu, 32, 16);
        tdm_load_tile_f32(slotUoff, pu + n0, (unsigned)ldgu, 32, 16);
#endif
        #pragma unroll 1
        for (int kk = 0; kk < HDIM; kk += 32) {
            v8h alo = *(const v8h*)(arow + kk + kB);
            v8h ahi = *(const v8h*)(arow + kk + kB + 8);
            v16h a  = concat8(alo, ahi);
            v16h bgv, buv;
#if USE_TDM
            const int cur = (kk >> 5) & 1;
            if (kk + 32 < HDIM) {          // prefetch next panel, drain current
                const unsigned boff = (unsigned)(cur ^ 1) * 2048u;
                tdm_load_tile_f32(slotGoff + boff,
                                  pg + (size_t)(kk + 32) * ldgu + n0,
                                  (unsigned)ldgu, 32, 16);
                tdm_load_tile_f32(slotUoff + boff,
                                  pu + (size_t)(kk + 32) * ldgu + n0,
                                  (unsigned)ldgu, 32, 16);
                TDM_WAIT(2);
            } else {
                TDM_WAIT(0);
            }
            const int gB = gBase0 + cur * 512 + kB * 16 + nIdx;
            const int uB = uBase0 + cur * 512 + kB * 16 + nIdx;
            #pragma unroll
            for (int j = 0; j < 16; ++j) {
                bgv[j] = (f16)SLOT[gB + j * 16];
                buv[j] = (f16)SLOT[uB + j * 16];
            }
#else
            const float* bg = pg + (size_t)(kk + kB) * ldgu + n0 + nIdx;
            const float* bu = pu + (size_t)(kk + kB) * ldgu + n0 + nIdx;
            #pragma unroll
            for (int j = 0; j < 16; ++j) {
                bgv[j] = (f16)bg[(size_t)j * ldgu];
                buv[j] = (f16)bu[(size_t)j * ldgu];
            }
#endif
            c1 = __builtin_amdgcn_wmma_f32_16x16x32_f16(false, a, false, bgv,
                                                        (short)0, c1, false, false);
            c2 = __builtin_amdgcn_wmma_f32_16x16x32_f16(false, a, false, buv,
                                                        (short)0, c2, false, false);
        }
        const int mrow = 8 * (lane >> 4);
        #pragma unroll
        for (int v = 0; v < 8; ++v) {
            float g = c1[v], u = c2[v];
            float sig = __builtin_amdgcn_rcpf(1.f + __expf(-g));  // fast sigmoid
            Hs[(mrow + v) * HS_STR + n0 + nIdx] = (f16)(g * sig * u);
        }
    }
    __syncthreads();

    // ---- Stage 3: out_tile = Hs @ Wd, scaled per-row, scatter atomicAdd -----
    for (int nt = wv; nt < HDIM / 16; nt += 8) {
        const int n0 = nt * 16;
        v8f d = {};
        const f16* arow = Hs + (lane & 15) * HS_STR;
#if USE_TDM
        tdm_load_tile_f32(slotGoff, pd + n0, HDIM, 32, 16);
#endif
        #pragma unroll 1
        for (int kk = 0; kk < IDIM; kk += 32) {
            v8h alo = *(const v8h*)(arow + kk + kB);
            v8h ahi = *(const v8h*)(arow + kk + kB + 8);
            v16h a  = concat8(alo, ahi);
            v16h bdv;
#if USE_TDM
            const int cur = (kk >> 5) & 1;
            if (kk + 32 < IDIM) {
                tdm_load_tile_f32(slotGoff + (unsigned)(cur ^ 1) * 2048u,
                                  pd + (size_t)(kk + 32) * HDIM + n0, HDIM, 32, 16);
                TDM_WAIT(1);
            } else {
                TDM_WAIT(0);
            }
            const int dB = gBase0 + cur * 512 + kB * 16 + nIdx;
            #pragma unroll
            for (int j = 0; j < 16; ++j) bdv[j] = (f16)SLOT[dB + j * 16];
#else
            const float* bd = pd + (size_t)(kk + kB) * HDIM + n0 + nIdx;
            #pragma unroll
            for (int j = 0; j < 16; ++j) bdv[j] = (f16)bd[(size_t)j * HDIM];
#endif
            d = __builtin_amdgcn_wmma_f32_16x16x32_f16(false, a, false, bdv,
                                                       (short)0, d, false, false);
        }
        const int mrow = 8 * (lane >> 4);
        #pragma unroll
        for (int v = 0; v < 8; ++v) {
            const int row = mrow + v;
            if (row < rowsValid) {
                float val = d[v] * Wt[row];
                atomicAdd(&out[(size_t)Ts[row] * HDIM + n0 + nIdx], val);
            }
        }
    }
}

// ---------------------------------------------------------------------------
// Host launcher
// ---------------------------------------------------------------------------
extern "C" void kernel_launch(void* const* d_in, const int* in_sizes, int n_in,
                              void* d_out, int out_size, void* d_ws, size_t ws_size,
                              hipStream_t stream) {
    (void)in_sizes; (void)n_in; (void)out_size; (void)ws_size;
    const float* x         = (const float*)d_in[0];
    const float* gate_w    = (const float*)d_in[1];
    const float* gate_bias = (const float*)d_in[2];
    const float* w_gate    = (const float*)d_in[3];
    const float* w_up      = (const float*)d_in[4];
    const float* w_down    = (const float*)d_in[5];
    const float* ws_gate   = (const float*)d_in[6];
    const float* ws_up     = (const float*)d_in[7];
    const float* ws_down   = (const float*)d_in[8];
    float* out = (float*)d_out;

    // Workspace: [cnt: 64 ints][exp_tok: 64*4096 ints][exp_w: 64*4096 floats]
    char* ws = (char*)d_ws;
    int*   cnt     = (int*)ws;
    int*   exp_tok = (int*)(ws + 1024);
    float* exp_w   = (float*)(ws + 1024 + (size_t)EXP * T_TOK * sizeof(int));

    hipMemsetAsync(cnt, 0, EXP * sizeof(int), stream);
    hipMemsetAsync(out, 0, (size_t)T_TOK * HDIM * sizeof(float), stream);

    router_kernel<<<dim3(T_TOK / 8), dim3(256), 0, stream>>>(
        x, gate_w, gate_bias, cnt, exp_tok, exp_w);

    moe_expert_kernel<<<dim3(T_TOK / 16, EXP + 2), dim3(256), LDS_TOTAL, stream>>>(
        x, w_gate, w_up, w_down, ws_gate, ws_up, ws_down,
        cnt, exp_tok, exp_w, out);
}
